// DynSMHALayer_6236292513902
// MI455X (gfx1250) — matrix operand order repaired and checked by reading, not compiled, not run on hardware
//
#include <hip/hip_runtime.h>
#include <hip/hip_bf16.h>
#include <math.h>

// Problem constants (match reference)
#define BB 4
#define TT 2048
#define CC 2048
#define HH 128
#define EE 16
#define NN (BB*TT)
#define NEGV (-1e9f)

typedef __attribute__((ext_vector_type(16))) _Float16 v16h;
typedef __attribute__((ext_vector_type(8)))  float    v8f;

union Frag { v16h v; unsigned u[8]; _Float16 h[16]; };

__device__ __forceinline__ v8f wmma16(v16h a, v16h b, v8f c) {
  // D = A(16x32 f16) * B(32x16 f16) + C(16x16 f32)
  return __builtin_amdgcn_wmma_f32_16x16x32_f16(false, a, false, b, (short)0, c, false, false);
}

__device__ __forceinline__ float rmax16(float v) {
  #pragma unroll
  for (int m = 1; m < 16; m <<= 1) v = fmaxf(v, __shfl_xor(v, m, 32));
  return v;
}
__device__ __forceinline__ float rsum16(float v) {
  #pragma unroll
  for (int m = 1; m < 16; m <<= 1) v += __shfl_xor(v, m, 32);
  return v;
}
__device__ __forceinline__ float rsum32(float v) {
  #pragma unroll
  for (int m = 1; m < 32; m <<= 1) v += __shfl_xor(v, m, 32);
  return v;
}

// ---------------------------------------------------------------------------
// Kernel 0: sim-matrix column inverse norms + sigmoid(gates)
// ---------------------------------------------------------------------------
__global__ void prep_kernel(const float* __restrict__ sim, const float* __restrict__ gates,
                            float* __restrict__ sninv, float* __restrict__ gb) {
  int e = threadIdx.x;
  if (e < EE) {
    float s = 0.f;
    for (int c = 0; c < CC; c++) { float t = sim[c * EE + e]; s += t * t; }
    sninv[e] = 1.0f / fmaxf(sqrtf(s), 1e-12f);
    gb[e] = 1.0f / (1.0f + __expf(-gates[e]));
  }
}

// ---------------------------------------------------------------------------
// Kernel 1: gating -> wmask[N,16]  (one wave32 per token)
// ---------------------------------------------------------------------------
__global__ __launch_bounds__(256) void gate_kernel(
    const float* __restrict__ x, const float* __restrict__ sim,
    const float* __restrict__ sninv, const float* __restrict__ gb,
    const int* __restrict__ minE, float* __restrict__ wmask) {
  int wave = threadIdx.x >> 5, lane = threadIdx.x & 31;
  int n = blockIdx.x * 8 + wave;

  float acc[EE];
  #pragma unroll
  for (int e = 0; e < EE; e++) acc[e] = 0.f;
  float sq = 0.f;
  const float* xr = x + (size_t)n * CC;
  for (int c = lane; c < CC; c += 32) {
    float xv = xr[c];
    sq += xv * xv;
    const float* srow = sim + c * EE;
    #pragma unroll
    for (int e = 0; e < EE; e++) acc[e] += xv * srow[e];
  }
  sq = rsum32(sq);
  float logit = -3.0e38f;
  #pragma unroll
  for (int e = 0; e < EE; e++) {
    float s = rsum32(acc[e]);
    if (lane == e) logit = s;
  }
  float xinv = 1.0f / fmaxf(sqrtf(sq), 1e-12f);
  if (lane < EE) logit = logit * xinv * sninv[lane] - gb[lane];

  bool active = (lane < EE) && (logit > 0.f);
  unsigned long long bal = __ballot(active);
  bool mask = active;
  if ((bal & 0xffffull) == 0ull) {
    int me = minE[0]; me = me < 1 ? 1 : (me > EE ? EE : me);
    bool sel = false;
    for (int t = 0; t < me; t++) {
      float bv = (lane < EE && !sel) ? logit : -3.0e38f;
      int bi = lane;
      #pragma unroll
      for (int off = 16; off; off >>= 1) {
        float ov = __shfl_xor(bv, off, 32);
        int   oi = __shfl_xor(bi, off, 32);
        if (ov > bv || (ov == bv && oi < bi)) { bv = ov; bi = oi; }
      }
      if (lane == bi) sel = true;
    }
    mask = sel;
  }
  float ml = (mask && lane < EE) ? fmaxf(logit, 0.f) : NEGV;
  float mx = rmax16(ml);
  float ex = __expf(ml - mx);
  float sum = rsum16(ex);
  float w = mask ? ex / sum : 0.f;
  if (lane < EE) wmask[n * EE + lane] = w;
}

// ---------------------------------------------------------------------------
// Kernel 2: pack a weight tensor W[E][Kdim][Ndim] (f32) into B-fragment-major
// f16 layout P[E][Ndim/16][Kdim/32][32 lanes][16 halves]:
//   lane = hi*16 + n_local, half hh -> k_local = hi*8 + (hh<8 ? hh : hh+8)
// ---------------------------------------------------------------------------
__global__ __launch_bounds__(256) void pack_w_kernel(
    const float* __restrict__ W, _Float16* __restrict__ P,
    int Kdim, int Ndim, int kcShift, int ntShift) {
  int idx = blockIdx.x * 256 + threadIdx.x;
  int hh = idx & 15;
  int lane = (idx >> 4) & 31;
  int t = idx >> 9;
  int kc = t & ((1 << kcShift) - 1); t >>= kcShift;
  int nt = t & ((1 << ntShift) - 1); t >>= ntShift;
  int e = t;
  int k = kc * 32 + (lane >> 4) * 8 + ((hh < 8) ? hh : hh + 8);
  int n = nt * 16 + (lane & 15);
  P[idx] = (_Float16)W[((size_t)e * Kdim + k) * Ndim + n];
}

// ---------------------------------------------------------------------------
// Kernel 3: expert-combined QKV projection via WMMA.
// grid.x = N/16 token tiles, grid.y = 0/1/2 (q/k/v). 8 waves * 16 cols = H.
// LDS: X tile in A-fragment-packed f16 layout [kc][lane][16 halves].
// ---------------------------------------------------------------------------
__global__ __launch_bounds__(256) void qkv_kernel(
    const float* __restrict__ x,
    const _Float16* __restrict__ wqp, const _Float16* __restrict__ wkp,
    const _Float16* __restrict__ wvp,
    const float* __restrict__ wmask,
    float* __restrict__ qo, float* __restrict__ ko, float* __restrict__ vo) {
  __shared__ __align__(64) _Float16 xt[64 * 512];  // 64 KB packed X tile
  const int tid = threadIdx.x;
  const int rowbase = blockIdx.x * 16;

  // fill LDS in A-fragment-packed layout (coalesced reads, scattered b16 LDS)
  #pragma unroll
  for (int i = 0; i < (16 * CC) / 256; i++) {
    int idx = i * 256 + tid;
    int r = idx >> 11, c = idx & (CC - 1);
    int kc = c >> 5, kl = c & 31;
    int hi2 = (kl >> 3) & 1;
    int hh = (kl & 7) + ((kl >> 4) << 3);
    int l2 = hi2 * 16 + r;
    xt[(kc << 9) + l2 * 16 + hh] = (_Float16)x[(size_t)(rowbase + r) * CC + c];
  }
  __syncthreads();

  const _Float16* Wp; float* O;
  if (blockIdx.y == 0)      { Wp = wqp; O = qo; }
  else if (blockIdx.y == 1) { Wp = wkp; O = ko; }
  else                      { Wp = wvp; O = vo; }

  int wave = tid >> 5, lane = tid & 31, ln = lane & 15, hi = lane >> 4;
  int ncol = wave * 16;

  v8f fin = {};
  for (int e = 0; e < EE; e++) {
    const _Float16* Wb = Wp + (((size_t)e * 8 + wave) * 64) * 512 + lane * 16;
    if (e + 1 < EE)
      __builtin_prefetch(Wp + (((size_t)(e + 1) * 8 + wave) * 64) * 512 + lane * 16, 0, 1);
    v8f a = {};
    for (int kc = 0; kc < 64; kc++) {
      Frag af, bf;
      af.v = *(const v16h*)(&xt[(kc << 9) + lane * 16]);
      bf.v = *(const v16h*)(Wb + (size_t)kc * 512);
      a = wmma16(af.v, bf.v, a);
    }
    // C/D layout: vgpr i -> row i (lanes 0-15) or i+8 (lanes 16-31)
    #pragma unroll
    for (int i = 0; i < 8; i++) {
      int row = rowbase + i + hi * 8;
      fin[i] += a[i] * wmask[row * EE + e];
    }
  }
  #pragma unroll
  for (int i = 0; i < 8; i++) {
    int row = rowbase + i + hi * 8;
    O[(size_t)row * HH + ncol + ln] = fin[i];
  }
}

// ---------------------------------------------------------------------------
// Kernel 4: repack q/k/v (f32 [N,H]) into fragment-packed f16 for attention.
// blockIdx.y: 0 = q (A-layout, 1/sqrt(H) folded), 1 = k (B-layout for S),
//             2 = v (B-layout for P@V).
// ---------------------------------------------------------------------------
__global__ __launch_bounds__(256) void repack_qkv_kernel(
    const float* __restrict__ q, const float* __restrict__ k, const float* __restrict__ v,
    _Float16* __restrict__ qp, _Float16* __restrict__ kp, _Float16* __restrict__ vp) {
  int idx = blockIdx.x * 256 + threadIdx.x;
  int hh = idx & 15;
  int lane = (idx >> 4) & 31;
  int hi = lane >> 4, ln = lane & 15;
  if (blockIdx.y == 0) {
    const float scale = 0.08838834764831845f;  // 1/sqrt(H)
    int hc = (idx >> 9) & 3;
    int qt = (idx >> 11) & 127;
    int b = idx >> 18;
    int row = b * TT + qt * 16 + ln;
    int h = hc * 32 + hi * 8 + ((hh < 8) ? hh : hh + 8);
    qp[idx] = (_Float16)(q[(size_t)row * HH + h] * scale);
  } else if (blockIdx.y == 1) {
    int hc = (idx >> 9) & 3;
    int kt = (idx >> 11) & 127;
    int b = idx >> 18;
    int row = b * TT + kt * 16 + ln;
    int h = hc * 32 + hi * 8 + ((hh < 8) ? hh : hh + 8);
    kp[idx] = (_Float16)k[(size_t)row * HH + h];
  } else {
    int ht = (idx >> 9) & 7;
    int kc2 = (idx >> 12) & 63;
    int b = idx >> 18;
    int row = b * TT + kc2 * 32 + hi * 8 + ((hh < 8) ? hh : hh + 8);
    int h = ht * 16 + ln;
    vp[idx] = (_Float16)v[(size_t)row * HH + h];
  }
}

// ---------------------------------------------------------------------------
// Kernel 5: causal flash attention, one wave32 per 16-query tile.
// All Q/K/V fragments are contiguous 32B loads from packed f16 buffers.
// ---------------------------------------------------------------------------
__global__ __launch_bounds__(256) void attn_kernel(
    const _Float16* __restrict__ qp, const _Float16* __restrict__ kp,
    const _Float16* __restrict__ vp, float* __restrict__ o) {
  __shared__ __align__(64) _Float16 pbuf[8 * 512];  // per-wave packed-P staging
  int wave = threadIdx.x >> 5, lane = threadIdx.x & 31, ln = lane & 15, hi = lane >> 4;
  int gw = blockIdx.x * 8 + wave;
  int batch = gw >> 7;          // TT/16 = 128 q-tiles per batch
  int qt = gw & 127;
  int qrow0 = batch * TT + qt * 16;
  _Float16* pw = pbuf + wave * 512;

  Frag qf[4];
  #pragma unroll
  for (int hc = 0; hc < 4; hc++)
    qf[hc].v = *(const v16h*)(qp + ((((size_t)batch * 128 + qt) * 4 + hc) << 9) + lane * 16);

  v8f acc[8];
  v8f z = {};
  #pragma unroll
  for (int t = 0; t < 8; t++) acc[t] = z;
  float mrow[8], lrow[8];
  #pragma unroll
  for (int i = 0; i < 8; i++) { mrow[i] = -3.0e38f; lrow[i] = 0.f; }

  int nkc = (qt * 16 + 47) >> 5;  // 32-key chunks covering causal extent
  for (int kc = 0; kc < nkc; kc++) {
    int kstart = kc * 32;
    int kt0 = kstart >> 4;

    v8f s0 = {}, s1 = {};
    #pragma unroll
    for (int hc = 0; hc < 4; hc++) {
      Frag b0, b1;
      b0.v = *(const v16h*)(kp + ((((size_t)batch * 128 + kt0)     * 4 + hc) << 9) + lane * 16);
      b1.v = *(const v16h*)(kp + ((((size_t)batch * 128 + kt0 + 1) * 4 + hc) << 9) + lane * 16);
      s0 = wmma16(qf[hc].v, b0.v, s0);
      s1 = wmma16(qf[hc].v, b1.v, s1);
    }

    float p0a[8], p1a[8];
    #pragma unroll
    for (int i = 0; i < 8; i++) {
      int qrow = qt * 16 + i + hi * 8;
      float sv0 = (kstart + ln      <= qrow) ? s0[i] : NEGV;
      float sv1 = (kstart + 16 + ln <= qrow) ? s1[i] : NEGV;
      float mv = rmax16(fmaxf(sv0, sv1));
      float mnew = fmaxf(mrow[i], mv);
      float sc = __expf(mrow[i] - mnew);
      mrow[i] = mnew;
      float e0 = __expf(sv0 - mnew);
      float e1 = __expf(sv1 - mnew);
      lrow[i] = lrow[i] * sc + rsum16(e0 + e1);
      #pragma unroll
      for (int t = 0; t < 8; t++) acc[t][i] *= sc;
      p0a[i] = e0; p1a[i] = e1;
    }

    // C-layout P -> A-fragment-packed LDS -> contiguous reload
    #pragma unroll
    for (int i = 0; i < 8; i++) {
      int lane0 = ((ln >> 3) & 1) * 16 + i + hi * 8;  // dest lane for cols ln / 16+ln
      int hb = ln & 7;
      pw[lane0 * 16 + hb]     = (_Float16)p0a[i];
      pw[lane0 * 16 + hb + 8] = (_Float16)p1a[i];
    }
    asm volatile("s_wait_dscnt 0x0" ::: "memory");
    Frag pf;
    pf.v = *(const v16h*)(&pw[lane * 16]);

    #pragma unroll
    for (int t = 0; t < 8; t++) {
      Frag vf;
      vf.v = *(const v16h*)(vp + ((((size_t)batch * 64 + kc) * 8 + t) << 9) + lane * 16);
      acc[t] = wmma16(pf.v, vf.v, acc[t]);
    }
  }

  #pragma unroll
  for (int i = 0; i < 8; i++) {
    float inv = 1.0f / lrow[i];
    int row = qrow0 + i + hi * 8;
    #pragma unroll
    for (int t = 0; t < 8; t++)
      o[(size_t)row * HH + t * 16 + ln] = acc[t][i] * inv;
  }
}

// ---------------------------------------------------------------------------
// Kernel 6: output projection  out = sum_e (attn * wmask_e) @ o_proj_e
// grid.x = N/16 token tiles, grid.y = C/128 col groups; wave per 16-col tile.
// ---------------------------------------------------------------------------
__global__ __launch_bounds__(256) void oproj_kernel(
    const float* __restrict__ attn, const float* __restrict__ wmask,
    const _Float16* __restrict__ wop, float* __restrict__ out) {
  __shared__ __align__(64) _Float16 atp[4 * 512];  // attn tile, A-packed f16
  __shared__ float wm[16 * EE];
  int tid = threadIdx.x;
  int rowbase = blockIdx.x * 16;

  #pragma unroll
  for (int i = 0; i < 8; i++) {
    int idx = i * 256 + tid;  // [hc][lane][hh]
    int hh = idx & 15;
    int lane2 = (idx >> 4) & 31;
    int hc = idx >> 9;
    int r = lane2 & 15;
    int h = hc * 32 + (lane2 >> 4) * 8 + ((hh < 8) ? hh : hh + 8);
    atp[idx] = (_Float16)attn[(size_t)(rowbase + r) * HH + h];
  }
  wm[tid & 255] = wmask[(rowbase + (tid >> 4)) * EE + (tid & 15)];
  __syncthreads();

  int wave = tid >> 5, lane = tid & 31, ln = lane & 15, hi = lane >> 4;
  int nt = blockIdx.y * 8 + wave;  // global 16-col tile index
  int nc = nt * 16;

  v8f acc = {};
  for (int e = 0; e < EE; e++) {
    float wl = (lane < EE) ? wm[lane * EE + e] : 0.f;
    if (__ballot(wl > 0.f) == 0ull) continue;   // expert inactive for whole tile
    _Float16 wh = (_Float16)wm[ln * EE + e];    // A-layout: lane -> row ln
    const _Float16* Wb = wop + ((((size_t)e * 128 + nt) * 4) << 9) + lane * 16;
    #pragma unroll
    for (int hc = 0; hc < 4; hc++) {
      Frag af, bf;
      af.v = (*(const v16h*)(&atp[(hc << 9) + lane * 16])) * wh;
      bf.v = *(const v16h*)(Wb + (hc << 9));
      acc = wmma16(af.v, bf.v, acc);
    }
  }
  #pragma unroll
  for (int i = 0; i < 8; i++) {
    int row = rowbase + i + hi * 8;
    out[(size_t)row * CC + nc + ln] = acc[i];
  }
}

// ---------------------------------------------------------------------------
extern "C" void kernel_launch(void* const* d_in, const int* in_sizes, int n_in,
                              void* d_out, int out_size, void* d_ws, size_t ws_size,
                              hipStream_t stream) {
  (void)in_sizes; (void)n_in; (void)out_size; (void)ws_size;
  const float* x   = (const float*)d_in[0];
  const float* sim = (const float*)d_in[1];
  const float* gts = (const float*)d_in[2];
  const float* qw  = (const float*)d_in[3];
  const float* kw  = (const float*)d_in[4];
  const float* vw  = (const float*)d_in[5];
  const float* ow  = (const float*)d_in[6];
  const int*   mE  = (const int*)d_in[7];
  float* out = (float*)d_out;

  float* ws    = (float*)d_ws;
  float* sninv = ws;                         // 16
  float* gb    = ws + 16;                    // 16 (pad to 64)
  float* wmsk  = ws + 64;                    // N*E
  float* qf32  = wmsk + (size_t)NN * EE;     // N*H f32
  float* kf32  = qf32 + (size_t)NN * HH;
  float* vf32  = kf32 + (size_t)NN * HH;
  float* ab    = vf32 + (size_t)NN * HH;
  float* half0 = ab + (size_t)NN * HH;
  _Float16* wqp = (_Float16*)half0;                      // E*C*H halves each
  _Float16* wkp = wqp + (size_t)EE * CC * HH;
  _Float16* wvp = wkp + (size_t)EE * CC * HH;
  _Float16* wop = wvp + (size_t)EE * CC * HH;
  _Float16* qp  = wop + (size_t)EE * CC * HH;            // N*H halves each
  _Float16* kp  = qp + (size_t)NN * HH;
  _Float16* vp  = kp + (size_t)NN * HH;                  // total ~57 MB

  prep_kernel<<<1, 32, 0, stream>>>(sim, gts, sninv, gb);
  gate_kernel<<<NN / 8, 256, 0, stream>>>(x, sim, sninv, gb, mE, wmsk);

  const int wtot = (EE * CC * HH) / 256;  // 16384 blocks per weight tensor
  pack_w_kernel<<<wtot, 256, 0, stream>>>(qw, wqp, CC, HH, 6, 3);
  pack_w_kernel<<<wtot, 256, 0, stream>>>(kw, wkp, CC, HH, 6, 3);
  pack_w_kernel<<<wtot, 256, 0, stream>>>(vw, wvp, CC, HH, 6, 3);
  pack_w_kernel<<<wtot, 256, 0, stream>>>(ow, wop, HH, CC, 2, 7);

  qkv_kernel<<<dim3(NN / 16, 3), 256, 0, stream>>>(x, wqp, wkp, wvp, wmsk,
                                                   qf32, kf32, vf32);
  repack_qkv_kernel<<<dim3((NN * HH) / 256, 3), 256, 0, stream>>>(
      qf32, kf32, vf32, qp, kp, vp);
  attn_kernel<<<(NN / 16) / 8, 256, 0, stream>>>(qp, kp, vp, ab);
  oproj_kernel<<<dim3(NN / 16, CC / 128), 256, 0, stream>>>(ab, wmsk, wop, out);
}
